// RangeAttention_2628519985690
// MI455X (gfx1250) — matrix operand order repaired and checked by reading, not compiled, not run on hardware
//
#include <hip/hip_runtime.h>
#include <hip/hip_bf16.h>

// ---------------------------------------------------------------------------
// Swin-style window attention block for MI455X (gfx1250, wave32, WMMA).
// All GEMMs + attention run on v_wmma_f32_16x16x32_f16 (f32 accumulate).
// GEMM tiles staged with GLOBAL_LOAD_ASYNC_TO_LDS_B128 (ASYNCcnt-tracked),
// double-buffered so the next tile's fill overlaps the current tile's WMMAs.
// Weights pre-transposed once so every tile fill is contiguous b128 traffic.
// Position-bias MLP is fused into the attention workgroup (never hits HBM).
// ---------------------------------------------------------------------------

#define C_DIM 256
#define HID_DIM 1024
#define L_DIM 32768
#define NH 8
#define HD 32
#define T_WIN 128
#define NWIN 256
#define SCALE_Q 0.17677669529663687f   // 32^-0.5
#define EPS_LN 1e-5f

typedef __attribute__((ext_vector_type(16))) _Float16 v16h;
typedef __attribute__((ext_vector_type(8)))  _Float16 v8h;
typedef __attribute__((ext_vector_type(8)))  float    v8f;

union V16 { v16h v; v8h h[2]; };

static __device__ __forceinline__ v8f wmma_f16(const V16& a, const V16& b, v8f c) {
    return __builtin_amdgcn_wmma_f32_16x16x32_f16(
        /*neg_a=*/false, a.v, /*neg_b=*/false, b.v,
        /*c_mod=*/(short)0, c, /*reuse_a=*/false, /*reuse_b=*/false);
}

// 16-byte async copy global -> LDS, tracked on ASYNCcnt.
// Flat LDS aperture: addr[31:0] of a generic pointer to __shared__ is the
// LDS byte offset, which is what VDST must carry.
static __device__ __forceinline__ void async_copy_b128(void* lds_dst, const void* gsrc) {
    uint32_t ldsoff = (uint32_t)(uintptr_t)lds_dst;
    uint64_t ga     = (uint64_t)(uintptr_t)gsrc;
    asm volatile("global_load_async_to_lds_b128 %0, %1, off"
                 :: "v"(ldsoff), "v"(ga) : "memory");
}
static __device__ __forceinline__ void wait_async0() {
    asm volatile("s_wait_asynccnt 0x0" ::: "memory");
}
// Drain all but the most recent 4 async ops (one pipeline stage).
static __device__ __forceinline__ void wait_async4() {
    asm volatile("s_wait_asynccnt 0x4" ::: "memory");
}

// token t of window 'win' -> flat row index l in [B,L,C]
static __device__ __forceinline__ int win_row(int win, int t) {
    // win = hb*16 + wb ; t = hi*32 + wi ; l = (hb*4+hi)*512 + wb*32 + wi
    return ((win >> 4) * 4 + (t >> 5)) * 512 + (win & 15) * 32 + (t & 31);
}

// ---------------------------------------------------------------------------
// fp32 -> f16 conversion with scale + transpose: src [K][N] -> dst [N][K].
// One-time cost; makes every GEMM B-tile fill contiguous.
// ---------------------------------------------------------------------------
__global__ void cvt_f16_t_kernel(const float* __restrict__ src,
                                 _Float16* __restrict__ dst,
                                 int K, int N, float scale) {
    int i = blockIdx.x * blockDim.x + threadIdx.x;
    if (i < K * N) {
        int k = i / N, n = i - k * N;
        dst[(size_t)n * K + k] = (_Float16)(src[i] * scale);
    }
}

// ---------------------------------------------------------------------------
// LayerNorm (C=256) -> f16 output.  One wave per row; shfl_xor reduction.
// ---------------------------------------------------------------------------
__global__ __launch_bounds__(256) void ln_f16_kernel(
    const float* __restrict__ x, const float* __restrict__ g,
    const float* __restrict__ b, _Float16* __restrict__ out) {
    int lane = threadIdx.x & 31, wv = threadIdx.x >> 5;
    int row = blockIdx.x * 8 + wv;
    const float* xr = x + (size_t)row * C_DIM;
    float v[8], s = 0.f, sq = 0.f;
#pragma unroll
    for (int i = 0; i < 8; ++i) {
        v[i] = xr[lane + 32 * i];
        s += v[i]; sq += v[i] * v[i];
    }
#pragma unroll
    for (int d = 16; d; d >>= 1) {
        s  += __shfl_xor(s, d, 32);
        sq += __shfl_xor(sq, d, 32);
    }
    float mean = s * (1.f / C_DIM);
    float var  = sq * (1.f / C_DIM) - mean * mean;
    float rs   = rsqrtf(var + EPS_LN);
    _Float16* orow = out + (size_t)row * C_DIM;
#pragma unroll
    for (int i = 0; i < 8; ++i) {
        int c = lane + 32 * i;
        orow[c] = (_Float16)((v[i] - mean) * rs * g[c] + b[c]);
    }
}

// ---------------------------------------------------------------------------
// Generic f16 WMMA GEMM:  out = A[M,K] @ W + bias, W given TRANSPOSED [N][K].
//   mode 0: store f16            mode 1: exact-erf GELU, store f16
//   mode 2: store f32 = acc + bias + resid
// 128x128 tile per block, 8 waves x (16 x 128) strip, K-steps of 32.
// Double-buffered async b128 tile fills: stage i+1 issued before waiting
// ASYNCcnt<=4 (in-order completion drains only stage i) -> fill latency
// hidden behind the 8 WMMAs of the current stage.
// ---------------------------------------------------------------------------
__global__ __launch_bounds__(256) void gemm_f16_kernel(
    const _Float16* __restrict__ A, const _Float16* __restrict__ WT,
    const float* __restrict__ bias, float bias_scale,
    const float* __restrict__ resid, void* __restrict__ outp,
    int M, int N, int K, int mode) {
    __shared__ _Float16 As[2][128][32];
    __shared__ _Float16 Bt[2][128][32];   // W tile, already transposed: [n][k]
    int tid = threadIdx.x, lane = tid & 31, wv = tid >> 5;
    int bm = blockIdx.x * 128, bn = blockIdx.y * 128;

    v8f acc[8];
#pragma unroll
    for (int i = 0; i < 8; ++i) acc[i] = (v8f)(0.f);

    int row = tid >> 1, c = (tid & 1) * 16;   // tile-fill assignment
    const _Float16* aptr = A  + (size_t)(bm + row) * K + c;
    const _Float16* bptr = WT + (size_t)(bn + row) * K + c;
    auto issue = [&](int buf, int kk) {
        async_copy_b128(&As[buf][row][c],     aptr + kk);
        async_copy_b128(&As[buf][row][c + 8], aptr + kk + 8);
        async_copy_b128(&Bt[buf][row][c],     bptr + kk);
        async_copy_b128(&Bt[buf][row][c + 8], bptr + kk + 8);
    };

    int nk = K >> 5;
    issue(0, 0);
    for (int i = 0; i < nk; ++i) {
        int cur = i & 1;
        if (i + 1 < nk) {             // prefetch next stage, then drain current
            issue(cur ^ 1, (i + 1) << 5);
            wait_async4();
        } else {
            wait_async0();
        }
        __syncthreads();              // current-stage LDS visible group-wide

        int arow = wv * 16 + (lane & 15);
        int kb   = (lane >> 4) * 8;     // A: K halves split 8/8 across lane halves
        V16 a;
        a.h[0] = *(const v8h*)&As[cur][arow][kb];
        a.h[1] = *(const v8h*)&As[cur][arow][kb + 16];
        int kb2 = (lane >> 4) * 8;      // B: K halves split 8/8 across lane halves
        int nl  = lane & 15;
#pragma unroll
        for (int nt = 0; nt < 8; ++nt) {
            V16 b;
            const _Float16* bp = &Bt[cur][nt * 16 + nl][kb2];
            b.h[0] = *(const v8h*)bp;
            b.h[1] = *(const v8h*)(bp + 16);
            acc[nt] = wmma_f16(a, b, acc[nt]);
        }
        __syncthreads();              // reads done before next overwrite of buf
    }

    // Epilogue.  D layout: vgpr r -> row r + 8*(lane/16); col = lane%16.
    int m_base = bm + wv * 16 + (lane >> 4) * 8;
    int nl = lane & 15;
#pragma unroll
    for (int nt = 0; nt < 8; ++nt) {
        int n = bn + nt * 16 + nl;
        float bv = bias[n] * bias_scale;
#pragma unroll
        for (int r = 0; r < 8; ++r) {
            size_t off = (size_t)(m_base + r) * N + n;
            float v = acc[nt][r] + bv;
            if (mode == 0) {
                ((_Float16*)outp)[off] = (_Float16)v;
            } else if (mode == 1) {
                float gl = 0.5f * v * (1.f + erff(v * 0.70710678118654752f));
                ((_Float16*)outp)[off] = (_Float16)gl;
            } else {
                ((float*)outp)[off] = v + resid[off];
            }
        }
    }
}

// ---------------------------------------------------------------------------
// Fused window attention: one block per (window, head).
//  - K tile (async b128) and V^T tile staged in LDS
//  - relative-position bias MLP (2->16->ReLU->1 per head) computed in-block
//  - S = bias + Q.K^T via WMMA (HD=32 -> exactly one wmma per 16x16 tile)
//  - in-LDS softmax, then O = P.V via WMMA (P converted f32->f16 on the fly)
// ---------------------------------------------------------------------------
__global__ __launch_bounds__(256) void attn_kernel(
    const _Float16* __restrict__ Q, const _Float16* __restrict__ Kb,
    const _Float16* __restrict__ Vb, const float* __restrict__ pos,
    const float* __restrict__ pw1, const float* __restrict__ pb1,
    const float* __restrict__ bn_g, const float* __restrict__ bn_b,
    const float* __restrict__ bn_m, const float* __restrict__ bn_v,
    const float* __restrict__ pw2, const float* __restrict__ pb2,
    _Float16* __restrict__ O) {
    __shared__ _Float16 k_s[T_WIN][HD];     // 8 KB
    __shared__ _Float16 vT[HD][T_WIN];      // 8 KB  (V transposed: [d][t])
    __shared__ float    S[T_WIN][T_WIN];    // 64 KB (bias -> logits -> probs)
    __shared__ float    px[T_WIN], py[T_WIN];

    int tid = threadIdx.x, lane = tid & 31, wv = tid >> 5;
    int win = blockIdx.x >> 3, head = blockIdx.x & 7;

    {   // stage K (async), V^T (register transpose), positions
        int t = tid >> 1, c = (tid & 1) * 16;
        size_t base = (size_t)win_row(win, t) * C_DIM + head * HD + c;
        async_copy_b128(&k_s[t][c],     Kb + base);
        async_copy_b128(&k_s[t][c + 8], Kb + base + 8);
        v8h v0 = *(const v8h*)(Vb + base);
        v8h v1 = *(const v8h*)(Vb + base + 8);
#pragma unroll
        for (int i = 0; i < 8; ++i) { vT[c + i][t] = v0[i]; vT[c + 8 + i][t] = v1[i]; }
        if (tid < T_WIN) {
            size_t lp = (size_t)win_row(win, tid) * 2;
            px[tid] = pos[lp];
            py[tid] = pos[lp + 1];
        }
        wait_async0();
    }
    __syncthreads();

    {   // relative position bias -> S  (BN folded into conv1)
        float ax[16], ay[16], cc[16], w2[16];
#pragma unroll
        for (int o = 0; o < 16; ++o) {
            float s = bn_g[o] * rsqrtf(bn_v[o] + EPS_LN);
            ax[o] = pw1[o * 2] * s;
            ay[o] = pw1[o * 2 + 1] * s;
            cc[o] = (pb1[o] - bn_m[o]) * s + bn_b[o];
            w2[o] = pw2[head * 16 + o];
        }
        float b2 = pb2[head];
        for (int e = tid; e < T_WIN * T_WIN; e += 256) {
            int t = e >> 7, u = e & 127;
            float rx = px[t] - px[u], ry = py[t] - py[u];
            float acc = b2;
#pragma unroll
            for (int o = 0; o < 16; ++o) {
                float h = fmaf(ax[o], rx, fmaf(ay[o], ry, cc[o]));
                acc = fmaf(w2[o], fmaxf(h, 0.f), acc);
            }
            S[t][u] = acc;
        }
    }
    __syncthreads();

    {   // S += Q . K^T  (per-wave 16x128 strip; one wmma per tile, K=HD=32)
        int m0 = wv * 16;
        int arow = m0 + (lane & 15);
        int kb  = (lane >> 4) * 8;
        int kb2 = (lane >> 4) * 8;
        int nl  = lane & 15;
        const _Float16* qp =
            Q + (size_t)win_row(win, arow) * C_DIM + head * HD;
        V16 a;
        a.h[0] = *(const v8h*)(qp + kb);
        a.h[1] = *(const v8h*)(qp + kb + 16);
        int rbase = m0 + (lane >> 4) * 8;
#pragma unroll
        for (int ut = 0; ut < 8; ++ut) {
            V16 b;   // B[k][n] = K[n][k] -> contiguous row of k_s
            const _Float16* bp = &k_s[ut * 16 + nl][kb2];
            b.h[0] = *(const v8h*)bp;
            b.h[1] = *(const v8h*)(bp + 16);
            v8f c;
#pragma unroll
            for (int r = 0; r < 8; ++r) c[r] = S[rbase + r][ut * 16 + nl];
            c = wmma_f16(a, b, c);
#pragma unroll
            for (int r = 0; r < 8; ++r) S[rbase + r][ut * 16 + nl] = c[r];
        }
    }
    __syncthreads();

    if (tid < T_WIN) {   // row softmax in LDS
        float mx = -1e30f;
#pragma unroll 4
        for (int u = 0; u < T_WIN; ++u) mx = fmaxf(mx, S[tid][u]);
        float sum = 0.f;
#pragma unroll 4
        for (int u = 0; u < T_WIN; ++u) {
            float e = __expf(S[tid][u] - mx);
            S[tid][u] = e;
            sum += e;
        }
        float inv = 1.f / sum;
#pragma unroll 4
        for (int u = 0; u < T_WIN; ++u) S[tid][u] *= inv;
    }
    __syncthreads();

    {   // O = P . V   (K=128 -> 4 wmma steps; 2 N-tiles of 16 cover HD=32)
        int m0 = wv * 16;
        int arow = m0 + (lane & 15);
        int kb  = (lane >> 4) * 8;
        int kb2 = (lane >> 4) * 8;
        int nl  = lane & 15;
        v8f acc0 = (v8f)(0.f), acc1 = (v8f)(0.f);
#pragma unroll
        for (int ks = 0; ks < T_WIN; ks += 32) {
            V16 a;
#pragma unroll
            for (int i = 0; i < 8; ++i) {
                a.v[i]     = (_Float16)S[arow][ks + kb + i];
                a.v[i + 8] = (_Float16)S[arow][ks + kb + 16 + i];
            }
            V16 b0, b1;   // B[k][d] = V[k][d] = vT[d][k] contiguous
            const _Float16* bp0 = &vT[nl][ks + kb2];
            b0.h[0] = *(const v8h*)bp0; b0.h[1] = *(const v8h*)(bp0 + 16);
            const _Float16* bp1 = &vT[16 + nl][ks + kb2];
            b1.h[0] = *(const v8h*)bp1; b1.h[1] = *(const v8h*)(bp1 + 16);
            acc0 = wmma_f16(a, b0, acc0);
            acc1 = wmma_f16(a, b1, acc1);
        }
        int rbase = m0 + (lane >> 4) * 8;
#pragma unroll
        for (int r = 0; r < 8; ++r) {
            int t = rbase + r;
            _Float16* op = O + (size_t)win_row(win, t) * C_DIM + head * HD;
            op[nl]      = (_Float16)acc0[r];
            op[16 + nl] = (_Float16)acc1[r];
        }
    }
}

// ---------------------------------------------------------------------------
extern "C" void kernel_launch(void* const* d_in, const int* in_sizes, int n_in,
                              void* d_out, int out_size, void* d_ws, size_t ws_size,
                              hipStream_t stream) {
    const float* x       = (const float*)d_in[0];
    const float* pos     = (const float*)d_in[1];
    const float* norm1_g = (const float*)d_in[2];
    const float* norm1_b = (const float*)d_in[3];
    const float* wq      = (const float*)d_in[4];
    const float* bq      = (const float*)d_in[5];
    const float* wk      = (const float*)d_in[6];
    const float* bk      = (const float*)d_in[7];
    const float* wvw     = (const float*)d_in[8];
    const float* bv      = (const float*)d_in[9];
    const float* wo      = (const float*)d_in[10];
    const float* bo      = (const float*)d_in[11];
    const float* norm2_g = (const float*)d_in[12];
    const float* norm2_b = (const float*)d_in[13];
    const float* w_fc1   = (const float*)d_in[14];
    const float* b_fc1   = (const float*)d_in[15];
    const float* w_fc2   = (const float*)d_in[16];
    const float* b_fc2   = (const float*)d_in[17];
    const float* pw1     = (const float*)d_in[18];
    const float* pb1     = (const float*)d_in[19];
    const float* bn_g    = (const float*)d_in[20];
    const float* bn_b    = (const float*)d_in[21];
    const float* bn_m    = (const float*)d_in[22];
    const float* bn_v    = (const float*)d_in[23];
    const float* pw2     = (const float*)d_in[24];
    const float* pb2     = (const float*)d_in[25];

    // ---- workspace layout (bytes) -----------------------------------------
    char* ws = (char*)d_ws;
    size_t off = 0;
    auto alloc = [&](size_t bytes) { char* p = ws + off; off += (bytes + 255) & ~(size_t)255; return p; };
    _Float16* wq16   = (_Float16*)alloc(C_DIM * C_DIM * 2);     // all transposed [N][K]
    _Float16* wk16   = (_Float16*)alloc(C_DIM * C_DIM * 2);
    _Float16* wv16   = (_Float16*)alloc(C_DIM * C_DIM * 2);
    _Float16* wo16   = (_Float16*)alloc(C_DIM * C_DIM * 2);
    _Float16* wfc1_16= (_Float16*)alloc(C_DIM * HID_DIM * 2);
    _Float16* wfc2_16= (_Float16*)alloc(HID_DIM * C_DIM * 2);
    _Float16* bufA   = (_Float16*)alloc((size_t)L_DIM * C_DIM * 2);   // xn, later ln2 out
    _Float16* bufQ   = (_Float16*)alloc((size_t)L_DIM * C_DIM * 2);
    _Float16* bufK   = (_Float16*)alloc((size_t)L_DIM * C_DIM * 2);
    _Float16* bufV   = (_Float16*)alloc((size_t)L_DIM * C_DIM * 2);
    _Float16* bufO   = (_Float16*)alloc((size_t)L_DIM * C_DIM * 2);
    float*    x1     = (float*)   alloc((size_t)L_DIM * C_DIM * 4);
    _Float16* bufG   = bufQ;  // fc1 output (L x 1024 f16 = 64MB) reuses Q..O span

    // ---- weight conversion + transpose (SCALE folded into wq) -------------
    int nw = C_DIM * C_DIM;
    int nf = C_DIM * HID_DIM;
    cvt_f16_t_kernel<<<(nw + 255) / 256, 256, 0, stream>>>(wq,    wq16,    C_DIM, C_DIM, SCALE_Q);
    cvt_f16_t_kernel<<<(nw + 255) / 256, 256, 0, stream>>>(wk,    wk16,    C_DIM, C_DIM, 1.f);
    cvt_f16_t_kernel<<<(nw + 255) / 256, 256, 0, stream>>>(wvw,   wv16,    C_DIM, C_DIM, 1.f);
    cvt_f16_t_kernel<<<(nw + 255) / 256, 256, 0, stream>>>(wo,    wo16,    C_DIM, C_DIM, 1.f);
    cvt_f16_t_kernel<<<(nf + 255) / 256, 256, 0, stream>>>(w_fc1, wfc1_16, C_DIM, HID_DIM, 1.f);
    cvt_f16_t_kernel<<<(nf + 255) / 256, 256, 0, stream>>>(w_fc2, wfc2_16, HID_DIM, C_DIM, 1.f);

    // ---- LN1 -> xn (f16) --------------------------------------------------
    ln_f16_kernel<<<L_DIM / 8, 256, 0, stream>>>(x, norm1_g, norm1_b, bufA);

    // ---- Q/K/V projections ------------------------------------------------
    dim3 gQKV(L_DIM / 128, C_DIM / 128);
    gemm_f16_kernel<<<gQKV, 256, 0, stream>>>(bufA, wq16, bq, SCALE_Q, nullptr,
                                              bufQ, L_DIM, C_DIM, C_DIM, 0);
    gemm_f16_kernel<<<gQKV, 256, 0, stream>>>(bufA, wk16, bk, 1.f, nullptr,
                                              bufK, L_DIM, C_DIM, C_DIM, 0);
    gemm_f16_kernel<<<gQKV, 256, 0, stream>>>(bufA, wv16, bv, 1.f, nullptr,
                                              bufV, L_DIM, C_DIM, C_DIM, 0);

    // ---- fused window attention (bias MLP + softmax + PV) -----------------
    attn_kernel<<<NWIN * NH, 256, 0, stream>>>(bufQ, bufK, bufV, pos,
                                               pw1, pb1, bn_g, bn_b, bn_m, bn_v,
                                               pw2, pb2, bufO);

    // ---- output projection + residual -> x1 (f32) -------------------------
    gemm_f16_kernel<<<gQKV, 256, 0, stream>>>(bufO, wo16, bo, 1.f, x,
                                              x1, L_DIM, C_DIM, C_DIM, 2);

    // ---- LN2 -> f16 (reuse bufA) ------------------------------------------
    ln_f16_kernel<<<L_DIM / 8, 256, 0, stream>>>(x1, norm2_g, norm2_b, bufA);

    // ---- FC1 + exact GELU -> bufG (f16) -----------------------------------
    dim3 gFC1(L_DIM / 128, HID_DIM / 128);
    gemm_f16_kernel<<<gFC1, 256, 0, stream>>>(bufA, wfc1_16, b_fc1, 1.f, nullptr,
                                              bufG, L_DIM, HID_DIM, C_DIM, 1);

    // ---- FC2 + residual -> d_out (f32) ------------------------------------
    dim3 gFC2(L_DIM / 128, C_DIM / 128);
    gemm_f16_kernel<<<gFC2, 256, 0, stream>>>(bufG, wfc2_16, b_fc2, 1.f, x1,
                                              (float*)d_out, L_DIM, C_DIM, HID_DIM, 2);
}
